// MultiHeadAttention_7722351199030
// MI455X (gfx1250) — matrix operand order repaired and checked by reading, not compiled, not run on hardware
//
#include <hip/hip_runtime.h>
#include <hip/hip_bf16.h>

#define SEQ    4096
#define DMODEL 1024
#define NHEADS 16
#define HDIM   64

typedef __attribute__((ext_vector_type(16))) __bf16 v16bf;
typedef __attribute__((ext_vector_type(8)))  float  v8f;
typedef __attribute__((ext_vector_type(4)))  unsigned int u32x4;
typedef __attribute__((ext_vector_type(4)))  int i32x4;
typedef __attribute__((ext_vector_type(8)))  int i32x8;

#if defined(__HIP_DEVICE_COMPILE__) && __has_builtin(__builtin_amdgcn_tensor_load_to_lds) && \
    __has_builtin(__builtin_amdgcn_s_wait_tensorcnt)
#define USE_TDM 1
#else
#define USE_TDM 0
#endif

union FragBF { v16bf v; uint4 q[2]; };
union FragF  { v8f  v; float f[8]; };

__device__ inline unsigned int pack_bf2(float a, float b) {
  union { __bf16 h[2]; unsigned int u; } p;
  p.h[0] = (__bf16)a; p.h[1] = (__bf16)b;
  return p.u;
}
__device__ inline unsigned short bf1(float a) {
  union { __bf16 h; unsigned short s; } p;
  p.h = (__bf16)a;
  return p.s;
}

__device__ inline v8f wmma_bf16(v16bf a, v16bf b, v8f c) {
#if defined(__HIP_DEVICE_COMPILE__)
  return __builtin_amdgcn_wmma_f32_16x16x32_bf16(false, a, false, b, (short)0, c, false, false);
#else
  return c;
#endif
}

// ---------------------------------------------------------------------------
// Tensor Data Mover: 2D bf16 tile (tile_rows x row_elems, element stride
// stride_elems between rows) from global -> LDS. One issue per wave.
// D# layout per CDNA5 ISA ch.8 (group0/group1; groups 2-3 zero for 2D).
// ---------------------------------------------------------------------------
__device__ inline void tdm_load_tile_2d(unsigned lds_off, const unsigned short* gptr,
                                        unsigned tile_rows, unsigned row_elems,
                                        unsigned long long stride_elems) {
#if USE_TDM
  unsigned long long ga = (unsigned long long)(uintptr_t)gptr;
  u32x4 g0;
  g0[0] = 1u;                                             // count=1 (valid), user D#
  g0[1] = lds_off;                                        // lds_addr (bytes)
  g0[2] = (unsigned)(ga & 0xffffffffu);                   // global_addr[31:0]
  g0[3] = (unsigned)((ga >> 32) & 0x01ffffffu) | (2u << 30); // addr[56:32] | type=2
  i32x8 g1;
  g1[0] = (int)(1u << 16);                      // wg_mask=0, data_size=1 (2 bytes)
  g1[1] = (int)(row_elems << 16);               // tensor_dim0 low16 (bits 63:48)
  g1[2] = 0;                                    // dim0 hi16=0 | tensor_dim1 low16=0
  g1[3] = (int)(16u | (row_elems << 16));       // tensor_dim1 hi16 (=1<<20) | tile_dim0
  g1[4] = (int)tile_rows;                       // tile_dim1 | tile_dim2=0
  g1[5] = (int)(unsigned)stride_elems;          // tensor_dim0_stride low32
  g1[6] = (int)(unsigned)(stride_elems >> 32);  // stride0 hi16 | stride1 low16=0
  g1[7] = 0;
  i32x4 z4 = {0, 0, 0, 0};
#if __clang_major__ >= 23
  i32x8 z8 = {0, 0, 0, 0, 0, 0, 0, 0};
  __builtin_amdgcn_tensor_load_to_lds(g0, g1, z4, z4, z8, 0);
#else
  __builtin_amdgcn_tensor_load_to_lds(g0, g1, z4, z4, 0);
#endif
#endif
}
__device__ inline unsigned lds_offset_of(const void* p) {
  return (unsigned)(size_t)p;   // low 32 bits of a shared pointer = LDS byte offset
}

// ---- shared GEMM helpers ---------------------------------------------------
__device__ inline void stage_f32_tile(const float* __restrict__ src, int m0, int k0,
                                      int tid, unsigned short (*lA)[32]) {
#pragma unroll
  for (int i = 0; i < 4; i++) {
    int e = tid + i * 256;
    int row = e >> 3, c4 = (e & 7) * 4;
    float4 f = *(const float4*)(src + (size_t)(m0 + row) * DMODEL + k0 + c4);
    uint2 p;
    p.x = pack_bf2(f.x, f.y);
    p.y = pack_bf2(f.z, f.w);
    *(uint2*)(&lA[row][c4]) = p;
  }
}
__device__ inline void stage_bf16_tile(const unsigned short* __restrict__ src, int m0,
                                       int k0, int tid, unsigned short (*lA)[32]) {
#pragma unroll
  for (int i = 0; i < 2; i++) {
    int e = tid + i * 256;
    int row = e >> 2, c8 = (e & 3) * 8;
    *(uint4*)(&lA[row][c8]) =
        *(const uint4*)(src + (size_t)(m0 + row) * DMODEL + k0 + c8);
  }
}
__device__ inline void stage_w_tile(const float* __restrict__ W, int n0, int k0,
                                    int tid, unsigned short (*lBt)[32]) {
#pragma unroll
  for (int i = 0; i < 2; i++) {
    int e = tid + i * 256;
    int kr = e >> 4, c4 = (e & 15) * 4;
    float4 f = *(const float4*)(W + (size_t)(k0 + kr) * DMODEL + n0 + c4);
    lBt[c4 + 0][kr] = bf1(f.x);
    lBt[c4 + 1][kr] = bf1(f.y);
    lBt[c4 + 2][kr] = bf1(f.z);
    lBt[c4 + 3][kr] = bf1(f.w);
  }
}
__device__ inline void mma_step(const unsigned short (*lA)[32],
                                const unsigned short (*lBt)[32],
                                int w, int lane, FragF* acc) {
  FragBF a;
  const int arow = w * 16 + (lane & 15);
  const int asel = (lane < 16) ? 0 : 8;
  a.q[0] = *(const uint4*)(&lA[arow][asel]);
  a.q[1] = *(const uint4*)(&lA[arow][asel + 16]);
#pragma unroll
  for (int nt = 0; nt < 4; nt++) {
    FragBF b;
    const int brow = nt * 16 + (lane & 15);
    const int bsel = (lane < 16) ? 0 : 8;
    b.q[0] = *(const uint4*)(&lBt[brow][bsel]);
    b.q[1] = *(const uint4*)(&lBt[brow][bsel + 16]);
    acc[nt].v = wmma_bf16(a.v, b.v, acc[nt].v);
  }
}

// ---------------------------------------------------------------------------
// Kernel 1: QKV projection, double-buffered LDS (f32 sources need conversion,
// so staging stays on the vector path).
// ---------------------------------------------------------------------------
__global__ __launch_bounds__(256) void qkv_gemm_kernel(
    const float* __restrict__ x,
    const float* __restrict__ Wq, const float* __restrict__ Wk, const float* __restrict__ Wv,
    unsigned short* __restrict__ Qb, unsigned short* __restrict__ Kb,
    unsigned short* __restrict__ Vt)
{
  __shared__ unsigned short lA[2][128][32];
  __shared__ unsigned short lBt[2][64][32];

  const int tid = threadIdx.x, w = tid >> 5, lane = tid & 31;
  const int m0 = blockIdx.y * 128, n0 = blockIdx.x * 64;
  const int z  = blockIdx.z;
  const float* W = (z == 0) ? Wq : ((z == 1) ? Wk : Wv);

  FragF acc[4];
#pragma unroll
  for (int t = 0; t < 4; t++)
#pragma unroll
    for (int r = 0; r < 8; r++) acc[t].f[r] = 0.f;

  stage_f32_tile(x, m0, 0, tid, lA[0]);
  stage_w_tile(W, n0, 0, tid, lBt[0]);

  int p = 0;
  for (int k0 = 0; k0 < DMODEL; k0 += 32) {
    __syncthreads();
    if (k0 + 32 < DMODEL) {
      stage_f32_tile(x, m0, k0 + 32, tid, lA[p ^ 1]);
      stage_w_tile(W, n0, k0 + 32, tid, lBt[p ^ 1]);
    }
    mma_step(lA[p], lBt[p], w, lane, acc);
    p ^= 1;
  }

  const int rbase = m0 + w * 16 + ((lane < 16) ? 0 : 8);
#pragma unroll
  for (int nt = 0; nt < 4; nt++) {
    const int col = n0 + nt * 16 + (lane & 15);
    const int head = col >> 6, dim = col & 63;
#pragma unroll
    for (int r = 0; r < 8; r++) {
      const int row = rbase + r;
      unsigned short hv = bf1(acc[nt].f[r]);
      if (z == 0)      Qb[((size_t)head * SEQ + row) * HDIM + dim] = hv;
      else if (z == 1) Kb[((size_t)head * SEQ + row) * HDIM + dim] = hv;
      else             Vt[((size_t)head * HDIM + dim) * SEQ + row] = hv;
    }
  }
}

// ---------------------------------------------------------------------------
// Kernel 2: causal flash attention; KV tiles staged by the Tensor Data Mover
// (wave 0 issues the DMA; TENSORcnt + workgroup barrier hand off the buffer).
// ---------------------------------------------------------------------------
__device__ inline void stage_kv(const unsigned short* __restrict__ Kb,
                                const unsigned short* __restrict__ Vt,
                                int head, int kbase, int tid,
                                unsigned short (*kb)[64], unsigned short (*vb)[64]) {
#pragma unroll
  for (int i = 0; i < 2; i++) {
    int e = tid + i * 256;
    int row = e >> 3, c8 = (e & 7) * 8;
    *(uint4*)(&kb[row][c8]) =
        *(const uint4*)(Kb + ((size_t)head * SEQ + kbase + row) * HDIM + c8);
    *(uint4*)(&vb[row][c8]) =
        *(const uint4*)(Vt + ((size_t)head * HDIM + row) * SEQ + kbase + c8);
  }
}

__global__ __launch_bounds__(256) void attn_kernel(
    const unsigned short* __restrict__ Qb, const unsigned short* __restrict__ Kb,
    const unsigned short* __restrict__ Vt, unsigned short* __restrict__ Ctx)
{
  __shared__ unsigned short kbuf[2][64][64];   // [key][d]
  __shared__ unsigned short vbuf[2][64][64];   // [d][key]
  __shared__ unsigned short pbuf[8][16][64];   // per-wave P [row][key]

  const int tid = threadIdx.x, w = tid >> 5, lane = tid & 31;
  const int head = blockIdx.y;
  const int q0 = blockIdx.x * 128;
  const int qrow = q0 + w * 16;

  FragBF aq[2];
  {
    const int myq = qrow + (lane & 15);
    const unsigned short* qptr = Qb + ((size_t)head * SEQ + myq) * HDIM;
#pragma unroll
    for (int s = 0; s < 2; s++) {
      const int base = s * 32;
      aq[s].q[0] = *(const uint4*)(qptr + base + ((lane < 16) ? 0 : 8));
      aq[s].q[1] = *(const uint4*)(qptr + base + ((lane < 16) ? 16 : 24));
    }
  }

  FragF o[4];
  float mrow[8], lrow[8];
#pragma unroll
  for (int t = 0; t < 4; t++)
#pragma unroll
    for (int r = 0; r < 8; r++) o[t].f[r] = 0.f;
#pragma unroll
  for (int r = 0; r < 8; r++) { mrow[r] = -INFINITY; lrow[r] = 0.f; }

  const float SC = 0.125f * 1.4426950408889634f;   // (1/sqrt(d)) * log2(e)
  const int jmax = (q0 + 127) >> 6;

#if USE_TDM
  if (w == 0) {
    tdm_load_tile_2d(lds_offset_of(&kbuf[0][0][0]),
                     Kb + ((size_t)head * SEQ) * HDIM, 64, 64, 64);
    tdm_load_tile_2d(lds_offset_of(&vbuf[0][0][0]),
                     Vt + (size_t)head * HDIM * SEQ, 64, 64, SEQ);
  }
#else
  stage_kv(Kb, Vt, head, 0, tid, kbuf[0], vbuf[0]);
#endif

  int p = 0;
  for (int j = 0; j <= jmax; j++) {
    const int kbase = j * 64;
#if USE_TDM
    if (w == 0) __builtin_amdgcn_s_wait_tensorcnt(0);
    __syncthreads();
    if (j < jmax && w == 0) {
      tdm_load_tile_2d(lds_offset_of(&kbuf[p ^ 1][0][0]),
                       Kb + ((size_t)head * SEQ + kbase + 64) * HDIM, 64, 64, 64);
      tdm_load_tile_2d(lds_offset_of(&vbuf[p ^ 1][0][0]),
                       Vt + (size_t)head * HDIM * SEQ + kbase + 64, 64, 64, SEQ);
    }
#else
    __syncthreads();
    if (j < jmax) stage_kv(Kb, Vt, head, kbase + 64, tid, kbuf[p ^ 1], vbuf[p ^ 1]);
#endif

    // S = Q K^T
    FragF s[4];
#pragma unroll
    for (int nt = 0; nt < 4; nt++) {
#pragma unroll
      for (int r = 0; r < 8; r++) s[nt].f[r] = 0.f;
      const int key = nt * 16 + (lane & 15);
#pragma unroll
      for (int sl = 0; sl < 2; sl++) {
        FragBF b;
        const int off = sl * 32 + ((lane < 16) ? 0 : 8);
        b.q[0] = *(const uint4*)(&kbuf[p][key][off]);
        b.q[1] = *(const uint4*)(&kbuf[p][key][off + 16]);
        s[nt].v = wmma_bf16(aq[sl].v, b.v, s[nt].v);
      }
    }

    // scale into log2 domain + causal mask
    const int qg = qrow + ((lane < 16) ? 0 : 8);
#pragma unroll
    for (int nt = 0; nt < 4; nt++) {
      const int kg = kbase + nt * 16 + (lane & 15);
#pragma unroll
      for (int r = 0; r < 8; r++) {
        float sv = s[nt].f[r] * SC;
        if (kg > qg + r) sv = -INFINITY;
        s[nt].f[r] = sv;
      }
    }

    // online softmax (row reductions within 16-lane halves)
#pragma unroll
    for (int r = 0; r < 8; r++) {
      float mx = fmaxf(fmaxf(s[0].f[r], s[1].f[r]), fmaxf(s[2].f[r], s[3].f[r]));
#pragma unroll
      for (int msk = 1; msk <= 8; msk <<= 1) mx = fmaxf(mx, __shfl_xor(mx, msk, 32));
      const float mnew = fmaxf(mrow[r], mx);
      const float alpha = exp2f(mrow[r] - mnew);
      mrow[r] = mnew;
      float ps = 0.f;
#pragma unroll
      for (int nt = 0; nt < 4; nt++) {
        float pv = exp2f(s[nt].f[r] - mnew);
        s[nt].f[r] = pv;
        ps += pv;
      }
#pragma unroll
      for (int msk = 1; msk <= 8; msk <<= 1) ps += __shfl_xor(ps, msk, 32);
      lrow[r] = lrow[r] * alpha + ps;
#pragma unroll
      for (int nt = 0; nt < 4; nt++) o[nt].f[r] *= alpha;
    }

    // P -> LDS (wave-private) -> A-operand layout; O += P V
#pragma unroll
    for (int nt = 0; nt < 4; nt++) {
      const int kk = nt * 16 + (lane & 15);
#pragma unroll
      for (int r = 0; r < 8; r++)
        pbuf[w][r + ((lane < 16) ? 0 : 8)][kk] = bf1(s[nt].f[r]);
    }
#pragma unroll
    for (int sl = 0; sl < 2; sl++) {
      FragBF ap;
      const unsigned short* prow = &pbuf[w][lane & 15][0];
      const int base = sl * 32;
      ap.q[0] = *(const uint4*)(prow + base + ((lane < 16) ? 0 : 8));
      ap.q[1] = *(const uint4*)(prow + base + ((lane < 16) ? 16 : 24));
#pragma unroll
      for (int nt = 0; nt < 4; nt++) {
        FragBF b;
        const int dcol = nt * 16 + (lane & 15);
        const int off = base + ((lane < 16) ? 0 : 8);
        b.q[0] = *(const uint4*)(&vbuf[p][dcol][off]);
        b.q[1] = *(const uint4*)(&vbuf[p][dcol][off + 16]);
        o[nt].v = wmma_bf16(ap.v, b.v, o[nt].v);
      }
    }
    p ^= 1;
  }

  // finalize: O /= l, write ctx bf16 [n][h*64+d]
#pragma unroll
  for (int nt = 0; nt < 4; nt++) {
    const int col = head * HDIM + nt * 16 + (lane & 15);
#pragma unroll
    for (int r = 0; r < 8; r++) {
      const int row = qrow + r + ((lane < 16) ? 0 : 8);
      Ctx[(size_t)row * DMODEL + col] = bf1(o[nt].f[r] * (1.f / lrow[r]));
    }
  }
}

// ---------------------------------------------------------------------------
// Kernel 3: output projection; A-tile (bf16 ctx) staged by TDM, W-tile by
// vector path (needs f32->bf16 conversion).
// ---------------------------------------------------------------------------
__global__ __launch_bounds__(256) void out_gemm_kernel(
    const unsigned short* __restrict__ Ctx, const float* __restrict__ Wo,
    const float* __restrict__ bo, float* __restrict__ out)
{
  __shared__ unsigned short lA[2][128][32];
  __shared__ unsigned short lBt[2][64][32];

  const int tid = threadIdx.x, w = tid >> 5, lane = tid & 31;
  const int m0 = blockIdx.y * 128, n0 = blockIdx.x * 64;

  FragF acc[4];
#pragma unroll
  for (int t = 0; t < 4; t++)
#pragma unroll
    for (int r = 0; r < 8; r++) acc[t].f[r] = 0.f;

#if USE_TDM
  if (w == 0)
    tdm_load_tile_2d(lds_offset_of(&lA[0][0][0]),
                     Ctx + (size_t)m0 * DMODEL, 128, 32, DMODEL);
#else
  stage_bf16_tile(Ctx, m0, 0, tid, lA[0]);
#endif
  stage_w_tile(Wo, n0, 0, tid, lBt[0]);

  int p = 0;
  for (int k0 = 0; k0 < DMODEL; k0 += 32) {
#if USE_TDM
    if (w == 0) __builtin_amdgcn_s_wait_tensorcnt(0);
#endif
    __syncthreads();
    if (k0 + 32 < DMODEL) {
#if USE_TDM
      if (w == 0)
        tdm_load_tile_2d(lds_offset_of(&lA[p ^ 1][0][0]),
                         Ctx + (size_t)m0 * DMODEL + k0 + 32, 128, 32, DMODEL);
#else
      stage_bf16_tile(Ctx, m0, k0 + 32, tid, lA[p ^ 1]);
#endif
      stage_w_tile(Wo, n0, k0 + 32, tid, lBt[p ^ 1]);
    }
    mma_step(lA[p], lBt[p], w, lane, acc);
    p ^= 1;
  }

  const int rbase = m0 + w * 16 + ((lane < 16) ? 0 : 8);
#pragma unroll
  for (int nt = 0; nt < 4; nt++) {
    const int col = n0 + nt * 16 + (lane & 15);
    const float bv = bo[col];
#pragma unroll
    for (int r = 0; r < 8; r++)
      out[(size_t)(rbase + r) * DMODEL + col] = acc[nt].f[r] + bv;
  }
}

// ---------------------------------------------------------------------------
extern "C" void kernel_launch(void* const* d_in, const int* in_sizes, int n_in,
                              void* d_out, int out_size, void* d_ws, size_t ws_size,
                              hipStream_t stream) {
  const float* x  = (const float*)d_in[0];
  const float* Wq = (const float*)d_in[1];
  const float* Wk = (const float*)d_in[2];
  const float* Wv = (const float*)d_in[3];
  const float* Wo = (const float*)d_in[4];
  const float* bo = (const float*)d_in[5];
  float* out = (float*)d_out;

  unsigned short* ws  = (unsigned short*)d_ws;
  const size_t mat = (size_t)SEQ * DMODEL;
  unsigned short* Qb  = ws;
  unsigned short* Kb  = ws + mat;
  unsigned short* Vt  = ws + 2 * mat;
  unsigned short* Ctx = ws + 3 * mat;

  qkv_gemm_kernel<<<dim3(DMODEL / 64, SEQ / 128, 3), 256, 0, stream>>>(
      x, Wq, Wk, Wv, Qb, Kb, Vt);
  attn_kernel<<<dim3(SEQ / 128, NHEADS), 256, 0, stream>>>(Qb, Kb, Vt, Ctx);
  out_gemm_kernel<<<dim3(DMODEL / 64, SEQ / 128), 256, 0, stream>>>(Ctx, Wo, bo, out);
}
